// FusedMultiHeadAttention_84997402788177
// MI455X (gfx1250) — compile-verified, matrix operand order checked by reading, not run on hardware
//
#include <hip/hip_runtime.h>
#include <math.h>

typedef __bf16 bf16;
typedef __attribute__((ext_vector_type(16))) __bf16 v16bf;
typedef __attribute__((ext_vector_type(8)))  float  v8f;
typedef __attribute__((ext_vector_type(4)))  unsigned int u32x4;
typedef __attribute__((ext_vector_type(8)))  int  i32x8;
typedef __attribute__((ext_vector_type(4)))  int  i32x4;

#define D_MODEL   1024
#define NUM_HEADS 16
#define HEAD_DIM  64
#define SEQ_T     2048
#define BATCH     2
#define M_TOT     (BATCH * SEQ_T)   /* 4096 */
#define N_QKV     (3 * D_MODEL)     /* 3072 */
#define INV_SCALE 0.125f            /* 1/sqrt(64) */

// A/B fragment: element e of the v16bf held by lane `lane` covers K index
// frag_k within the 32-wide K chunk (CDNA5 ISA 16-bit A 16x32 layout).
__device__ __forceinline__ int frag_k(int lane, int e) {
    int v = e >> 1;
    return ((v & 3) << 1) + ((v >> 2) << 4) + ((lane >> 4) << 3) + (e & 1);
}

// ---------------------------------------------------------------------------
// LDS staging helpers (256-thread blocks), 32-wide K panels, stride 40.
// ---------------------------------------------------------------------------
template <int P>
__device__ __forceinline__ void stage_f32(bf16 (*dst)[40],
                                          const float* __restrict__ src, int tid) {
#pragma unroll
    for (int p = 0; p < P; ++p) {
        int idx = (p * 256 + tid) << 2;
        int row = idx >> 5, col = idx & 31;
        float4 v = *(const float4*)(src + (size_t)row * D_MODEL + col);
        dst[row][col + 0] = (bf16)v.x; dst[row][col + 1] = (bf16)v.y;
        dst[row][col + 2] = (bf16)v.z; dst[row][col + 3] = (bf16)v.w;
    }
}

template <int P>
__device__ __forceinline__ void stage_bf16(bf16 (*dst)[40],
                                           const bf16* __restrict__ src, int tid) {
#pragma unroll
    for (int p = 0; p < P; ++p) {
        int idx = (p * 256 + tid) << 3;
        int row = idx >> 5, col = idx & 31;
        *(uint4*)&dst[row][col] = *(const uint4*)(src + (size_t)row * D_MODEL + col);
    }
}

// ---------------------------------------------------------------------------
// TDM: async-DMA a 64-row x 128B tile (row stride `stride_units` 8-byte units)
// into LDS with 16B padding per 128B row -> padded LDS stride 72 bf16 (144B).
// Descriptor per CDNA5 ISA 8.3/8.4: 2D tile of 16x64 8-byte units.
// ---------------------------------------------------------------------------
__device__ __forceinline__ void tdm_load_tile64(const bf16* gsrc, unsigned lds_byte_off,
                                                unsigned stride_units, unsigned rows_total) {
    unsigned long long ga = (unsigned long long)(uintptr_t)gsrc;
    u32x4 g0;
    g0[0] = 1u;                                   // count=1, user descriptor
    g0[1] = lds_byte_off;                         // lds_addr
    g0[2] = (unsigned)(ga & 0xffffffffu);         // global_addr[31:0]
    g0[3] = (unsigned)((ga >> 32) & 0x01ffffffu)  // global_addr[56:32]
            | 0x80000000u;                        // type=2 ("image")
    i32x8 g1;
    // data_size=3 (8B) | pad_enable | pad_interval=4 (32 DW=128B) | pad_amount=3 (4 DW=16B)
    g1[0] = (3 << 16) | (1 << 20) | (4 << 22) | (3 << 25);
    g1[1] = (int)((stride_units & 0xffffu) << 16);        // tensor_dim0 lo16
    g1[2] = (int)((stride_units >> 16) & 0xffffu)         // tensor_dim0 hi16
          | (int)((rows_total & 0xffffu) << 16);          // tensor_dim1 lo16
    g1[3] = (int)((rows_total >> 16) & 0xffffu)           // tensor_dim1 hi16
          | (16 << 16);                                   // tile_dim0 = 16 units (128B)
    g1[4] = 64;                                           // tile_dim1 = 64 rows; tile_dim2=0
    g1[5] = (int)stride_units;                            // tensor_dim0_stride lo32
    g1[6] = 0;                                            // stride hi16 | dim1_stride lo16
    g1[7] = 0;
    i32x4 z4 = {0, 0, 0, 0};
#if defined(__clang_major__) && (__clang_major__ >= 23)
    i32x8 z8 = {0, 0, 0, 0, 0, 0, 0, 0};
    __builtin_amdgcn_tensor_load_to_lds(g0, g1, z4, z4, z8, 0);
#else
    __builtin_amdgcn_tensor_load_to_lds(g0, g1, z4, z4, 0);
#endif
}

// ---------------------------------------------------------------------------
// Kernel 1: qkv = x @ W_qkv^T + b_qkv; scatters Q,K head-major [B,H,T,d] and
// V transposed [B,H,d,T] (so attention's P@V B-operand is contiguous in K).
// 128x64 block tile, 8 waves, double-buffered LDS, 4 WMMAs per wave per K-step.
// ---------------------------------------------------------------------------
__global__ __launch_bounds__(256)
void qkv_gemm_kernel(const float* __restrict__ X,     // [4096, 1024]
                     const float* __restrict__ W,     // [3072, 1024]
                     const float* __restrict__ bias,  // [3072]
                     bf16* __restrict__ Qb, bf16* __restrict__ Kb,
                     bf16* __restrict__ Vt)
{
    __shared__ __align__(16) bf16 As[2][128][40];
    __shared__ __align__(16) bf16 Bs[2][64][40];

    const int tid  = threadIdx.x;
    const int wv   = tid >> 5;
    const int lane = tid & 31;
    const int mr   = lane & 15;
    const int hh   = lane >> 4;
    const int m0   = blockIdx.y * 128;
    const int n0   = blockIdx.x * 64;

    const v8f vzero = {0.f,0.f,0.f,0.f,0.f,0.f,0.f,0.f};
    v8f acc[4];
#pragma unroll
    for (int i = 0; i < 4; ++i) acc[i] = vzero;

    int kidx[16];
#pragma unroll
    for (int e = 0; e < 16; ++e) kidx[e] = frag_k(lane, e);

    stage_f32<4>(As[0], X + (size_t)m0 * D_MODEL, tid);
    stage_f32<2>(Bs[0], W + (size_t)n0 * D_MODEL, tid);

    int buf = 0;
    for (int kk = 0; kk < D_MODEL; kk += 32) {
        __syncthreads();
        if (kk + 32 < D_MODEL) {
            stage_f32<4>(As[buf ^ 1], X + (size_t)m0 * D_MODEL + kk + 32, tid);
            stage_f32<2>(Bs[buf ^ 1], W + (size_t)n0 * D_MODEL + kk + 32, tid);
            if (kk + 64 < D_MODEL) {
                __builtin_prefetch(X + (size_t)(m0 + (tid >> 1)) * D_MODEL + kk + 64, 0, 0);
                __builtin_prefetch(W + (size_t)(n0 + (tid >> 2)) * D_MODEL + kk + 64, 0, 0);
            }
        }
        v16bf a;
#pragma unroll
        for (int e = 0; e < 16; ++e) a[e] = As[buf][wv * 16 + mr][kidx[e]];
        v16bf bfr[4];
#pragma unroll
        for (int nt = 0; nt < 4; ++nt)
#pragma unroll
            for (int e = 0; e < 16; ++e) bfr[nt][e] = Bs[buf][nt * 16 + mr][kidx[e]];
#pragma unroll
        for (int nt = 0; nt < 4; ++nt)
            acc[nt] = __builtin_amdgcn_wmma_f32_16x16x32_bf16(
                false, a, false, bfr[nt], (short)0, acc[nt], false, false);
        buf ^= 1;
    }

    // Epilogue: sel is block-uniform (n0 is 64-aligned, Q/K/V bounds 1024-aligned)
    const int sel = n0 >> 10;                      // 0=Q,1=K,2=V (scalar)
    const float scl = (sel == 0) ? INV_SCALE : 1.0f;
    if (sel == 2) {                                 // V: write transposed [B,H,d,T]
#pragma unroll
        for (int nt = 0; nt < 4; ++nt) {
            int n    = n0 + nt * 16 + mr;
            float bv = bias[n];
            int rem  = n & 1023;
            int head = rem >> 6;
            int dim  = rem & 63;
#pragma unroll
            for (int r = 0; r < 8; ++r) {
                int m  = m0 + wv * 16 + r + (hh << 3);
                int bb = m >> 11;
                int t  = m & (SEQ_T - 1);
                float val = acc[nt][r] + bv;
                size_t dst = (((size_t)bb * NUM_HEADS + head) * HEAD_DIM + dim) * SEQ_T + t;
                Vt[dst] = (bf16)val;
            }
        }
    } else {                                        // Q/K: head-major [B,H,T,d]
        bf16* const dstBuf = (sel == 0) ? Qb : Kb;
#pragma unroll
        for (int nt = 0; nt < 4; ++nt) {
            int n    = n0 + nt * 16 + mr;
            float bv = bias[n];
            int rem  = n & 1023;
            int head = rem >> 6;
            int dim  = rem & 63;
#pragma unroll
            for (int r = 0; r < 8; ++r) {
                int m  = m0 + wv * 16 + r + (hh << 3);
                int bb = m >> 11;
                int t  = m & (SEQ_T - 1);
                float val = (acc[nt][r] + bv) * scl;
                size_t dst = (((size_t)bb * NUM_HEADS + head) * SEQ_T + t) * HEAD_DIM + dim;
                dstBuf[dst] = (bf16)val;
            }
        }
    }
}

// ---------------------------------------------------------------------------
// Kernel 2: flash attention. Grid (T/64, B*H), 4 waves, each wave = 16 Q rows.
// K (key-major) and V^T (d-major) tiles streamed by the Tensor Data Mover,
// double-buffered in LDS; all WMMA fragments are contiguous b128 LDS loads.
// ---------------------------------------------------------------------------
__global__ __launch_bounds__(128)
void attn_kernel(const bf16* __restrict__ Qb, const bf16* __restrict__ Kb,
                 const bf16* __restrict__ Vt, bf16* __restrict__ AO)
{
    __shared__ __align__(16) bf16 Ksm[2][64][72];   // [key][d]
    __shared__ __align__(16) bf16 Vsm[2][64][72];   // [d][key]
    __shared__ __align__(16) bf16 Psm[4][16][72];

    const int tid  = threadIdx.x;
    const int wv   = tid >> 5;
    const int lane = tid & 31;
    const int mr   = lane & 15;
    const int hh   = lane >> 4;
    const int bh   = blockIdx.y;
    const int bI   = bh / NUM_HEADS;
    const int head = bh % NUM_HEADS;
    const int q0   = blockIdx.x * 64;

    const bf16* Kbase  = Kb + (size_t)bh * SEQ_T * HEAD_DIM;   // [T][d]
    const bf16* Vtbase = Vt + (size_t)bh * HEAD_DIM * SEQ_T;   // [d][T]

    // Kick off TDM for tile 0 (wave 0 issues; TENSORcnt tracks completion)
    if (wv == 0) {
        tdm_load_tile64(Kbase, (unsigned)(uintptr_t)&Ksm[0][0][0],
                        HEAD_DIM * 2 / 8, SEQ_T);
        tdm_load_tile64(Vtbase, (unsigned)(uintptr_t)&Vsm[0][0][0],
                        SEQ_T * 2 / 8, HEAD_DIM);
    }

    int kidx[16];
#pragma unroll
    for (int e = 0; e < 16; ++e) kidx[e] = frag_k(lane, e);

    // Q fragments (loaded once from global while TDM streams tile 0)
    const bf16* Qrow = Qb + ((size_t)bh * SEQ_T + q0 + wv * 16 + mr) * HEAD_DIM;
    v16bf qa[2];
#pragma unroll
    for (int kc = 0; kc < 2; ++kc)
#pragma unroll
        for (int e = 0; e < 16; ++e) qa[kc][e] = Qrow[kc * 32 + kidx[e]];

    const v8f vzero = {0.f,0.f,0.f,0.f,0.f,0.f,0.f,0.f};
    v8f o[4];
#pragma unroll
    for (int i = 0; i < 4; ++i) o[i] = vzero;
    float mrun[8], lrun[8];
#pragma unroll
    for (int r = 0; r < 8; ++r) { mrun[r] = -INFINITY; lrun[r] = 0.f; }

    int buf = 0;
    for (int kt = 0; kt < SEQ_T / 64; ++kt) {
        if (wv == 0) __builtin_amdgcn_s_wait_tensorcnt(0);
        __syncthreads();                 // current buffer staged for everyone
        if (wv == 0 && kt + 1 < SEQ_T / 64) {
            tdm_load_tile64(Kbase + (size_t)(kt + 1) * 64 * HEAD_DIM,
                            (unsigned)(uintptr_t)&Ksm[buf ^ 1][0][0],
                            HEAD_DIM * 2 / 8, SEQ_T);
            tdm_load_tile64(Vtbase + (size_t)(kt + 1) * 64,   // column offset in V^T
                            (unsigned)(uintptr_t)&Vsm[buf ^ 1][0][0],
                            SEQ_T * 2 / 8, HEAD_DIM);
        }

        // S = Q @ K^T  (16 x 64 tile per wave)
        v8f s[4];
#pragma unroll
        for (int i = 0; i < 4; ++i) s[i] = vzero;
#pragma unroll
        for (int nt = 0; nt < 4; ++nt) {
#pragma unroll
            for (int kc = 0; kc < 2; ++kc) {
                v16bf bk;
#pragma unroll
                for (int e = 0; e < 16; ++e)
                    bk[e] = Ksm[buf][nt * 16 + mr][kc * 32 + kidx[e]];
                s[nt] = __builtin_amdgcn_wmma_f32_16x16x32_bf16(
                    false, qa[kc], false, bk, (short)0, s[nt], false, false);
            }
        }

        // Online softmax (rows m = r + 8*hh; cols spread over 16-lane halves)
#pragma unroll
        for (int r = 0; r < 8; ++r) {
            float lm = fmaxf(fmaxf(s[0][r], s[1][r]), fmaxf(s[2][r], s[3][r]));
#pragma unroll
            for (int off = 8; off >= 1; off >>= 1)
                lm = fmaxf(lm, __shfl_xor(lm, off, 32));
            float mnew  = fmaxf(mrun[r], lm);
            float alpha = __expf(mrun[r] - mnew);
            mrun[r] = mnew;
            float rs = 0.f;
#pragma unroll
            for (int nt = 0; nt < 4; ++nt) {
                float pv = __expf(s[nt][r] - mnew);
                s[nt][r] = pv;
                rs += pv;
            }
#pragma unroll
            for (int off = 8; off >= 1; off >>= 1)
                rs += __shfl_xor(rs, off, 32);
            lrun[r] = lrun[r] * alpha + rs;
#pragma unroll
            for (int nt = 0; nt < 4; ++nt) o[nt][r] *= alpha;
        }

        // P (C-layout) -> LDS -> A-layout transpose, wave-local (LDS in-order)
#pragma unroll
        for (int nt = 0; nt < 4; ++nt)
#pragma unroll
            for (int r = 0; r < 8; ++r)
                Psm[wv][r + (hh << 3)][nt * 16 + mr] = (bf16)s[nt][r];

        // O += P @ V   (B-operand from V^T: rows = d, contiguous keys)
#pragma unroll
        for (int kc = 0; kc < 2; ++kc) {
            v16bf pa;
#pragma unroll
            for (int e = 0; e < 16; ++e)
                pa[e] = Psm[wv][mr][kc * 32 + kidx[e]];
#pragma unroll
            for (int nt = 0; nt < 4; ++nt) {
                v16bf bv;
#pragma unroll
                for (int e = 0; e < 16; ++e)
                    bv[e] = Vsm[buf][nt * 16 + mr][kc * 32 + kidx[e]];
                o[nt] = __builtin_amdgcn_wmma_f32_16x16x32_bf16(
                    false, pa, false, bv, (short)0, o[nt], false, false);
            }
        }
        buf ^= 1;
    }

    // Normalize and write bf16 attn-out already re-transposed to [B,T,C]
#pragma unroll
    for (int r = 0; r < 8; ++r) {
        int t = q0 + wv * 16 + r + (hh << 3);
        float inv = 1.0f / lrun[r];
#pragma unroll
        for (int nt = 0; nt < 4; ++nt) {
            float val = o[nt][r] * inv;
            AO[((size_t)bI * SEQ_T + t) * D_MODEL + head * HEAD_DIM + nt * 16 + mr] =
                (bf16)val;
        }
    }
}

// ---------------------------------------------------------------------------
// Kernel 3: out = AO(bf16) @ W_o^T + b_o -> f32. Same tiling as kernel 1.
// ---------------------------------------------------------------------------
__global__ __launch_bounds__(256)
void out_gemm_kernel(const bf16* __restrict__ A,     // [4096, 1024] bf16
                     const float* __restrict__ W,    // [1024, 1024]
                     const float* __restrict__ bias, // [1024]
                     float* __restrict__ out)        // [4096, 1024]
{
    __shared__ __align__(16) bf16 As[2][128][40];
    __shared__ __align__(16) bf16 Bs[2][64][40];

    const int tid  = threadIdx.x;
    const int wv   = tid >> 5;
    const int lane = tid & 31;
    const int mr   = lane & 15;
    const int hh   = lane >> 4;
    const int m0   = blockIdx.y * 128;
    const int n0   = blockIdx.x * 64;

    const v8f vzero = {0.f,0.f,0.f,0.f,0.f,0.f,0.f,0.f};
    v8f acc[4];
#pragma unroll
    for (int i = 0; i < 4; ++i) acc[i] = vzero;

    int kidx[16];
#pragma unroll
    for (int e = 0; e < 16; ++e) kidx[e] = frag_k(lane, e);

    stage_bf16<2>(As[0], A + (size_t)m0 * D_MODEL, tid);
    stage_f32<2>(Bs[0], W + (size_t)n0 * D_MODEL, tid);

    int buf = 0;
    for (int kk = 0; kk < D_MODEL; kk += 32) {
        __syncthreads();
        if (kk + 32 < D_MODEL) {
            stage_bf16<2>(As[buf ^ 1], A + (size_t)m0 * D_MODEL + kk + 32, tid);
            stage_f32<2>(Bs[buf ^ 1], W + (size_t)n0 * D_MODEL + kk + 32, tid);
            if (kk + 64 < D_MODEL) {
                __builtin_prefetch(A + (size_t)(m0 + (tid >> 1)) * D_MODEL + kk + 64, 0, 0);
                __builtin_prefetch(W + (size_t)(n0 + (tid >> 2)) * D_MODEL + kk + 64, 0, 0);
            }
        }
        v16bf a;
#pragma unroll
        for (int e = 0; e < 16; ++e) a[e] = As[buf][wv * 16 + mr][kidx[e]];
        v16bf bfr[4];
#pragma unroll
        for (int nt = 0; nt < 4; ++nt)
#pragma unroll
            for (int e = 0; e < 16; ++e) bfr[nt][e] = Bs[buf][nt * 16 + mr][kidx[e]];
#pragma unroll
        for (int nt = 0; nt < 4; ++nt)
            acc[nt] = __builtin_amdgcn_wmma_f32_16x16x32_bf16(
                false, a, false, bfr[nt], (short)0, acc[nt], false, false);
        buf ^= 1;
    }

#pragma unroll
    for (int nt = 0; nt < 4; ++nt) {
        int n    = n0 + nt * 16 + mr;
        float bv = bias[n];
#pragma unroll
        for (int r = 0; r < 8; ++r) {
            int m = m0 + wv * 16 + r + (hh << 3);
            out[(size_t)m * D_MODEL + n] = acc[nt][r] + bv;
        }
    }
}

// ---------------------------------------------------------------------------
extern "C" void kernel_launch(void* const* d_in, const int* in_sizes, int n_in,
                              void* d_out, int out_size, void* d_ws, size_t ws_size,
                              hipStream_t stream) {
    (void)in_sizes; (void)n_in; (void)out_size; (void)ws_size;
    const float* x    = (const float*)d_in[0];
    const float* Wqkv = (const float*)d_in[1];
    const float* bqkv = (const float*)d_in[2];
    const float* Wo   = (const float*)d_in[3];
    const float* bo   = (const float*)d_in[4];
    float* out = (float*)d_out;

    // Workspace: Q,K [B,H,T,d] + V^T [B,H,d,T] + attn-out [B,T,C], all bf16 = 32 MB
    bf16* Qb = (bf16*)d_ws;
    bf16* Kb = Qb + (size_t)M_TOT * D_MODEL;
    bf16* Vt = Kb + (size_t)M_TOT * D_MODEL;
    bf16* AO = Vt + (size_t)M_TOT * D_MODEL;

    dim3 g1(N_QKV / 64, M_TOT / 128);          // (48, 32)
    qkv_gemm_kernel<<<g1, dim3(256), 0, stream>>>(x, Wqkv, bqkv, Qb, Kb, Vt);

    dim3 g2(SEQ_T / 64, BATCH * NUM_HEADS);    // (32, 32)
    attn_kernel<<<g2, dim3(128), 0, stream>>>(Qb, Kb, Vt, AO);

    dim3 g3(D_MODEL / 64, M_TOT / 128);        // (16, 32)
    out_gemm_kernel<<<g3, dim3(256), 0, stream>>>(AO, Wo, bo, out);
}
